// HATBlock_63548336111769
// MI455X (gfx1250) — matrix operand order repaired and checked by reading, not compiled, not run on hardware
//
#include <hip/hip_runtime.h>
#include <stdint.h>

typedef __attribute__((ext_vector_type(16))) _Float16 v16h;
typedef __attribute__((ext_vector_type(8)))  float    v8f;

union FragH { v16h h; unsigned int u[8]; };

#define T_TOK   147456   // B*H*W tokens
#define NWIN    2304     // 4 * 24 * 24 windows
#define CDIM    256
#define NHEAD   8
#define HIMG    192
#define WIMG    192
#define HWIMG   36864

// ---- scratch layout (bytes) ----
#define OFF_WQKV   0u                    // 98304 dwords
#define OFF_WPROJ  393216u               // 32768 dwords
#define OFF_W1     524288u               // 131072 dwords
#define OFF_W2     1048576u              // 131072 dwords
#define OFF_POOL   1572864u              // 1024 f32
#define OFF_G      1576960u              // 1024 f32
#define OFF_BTAB   1581056u              // 8*64*64 f32 = 131072 B
#define OFF_H      2097152u              // h (f16, 75497472 B) -> o -> h3 (reused)
#define OFF_BIG    77594624u             // q,k,v (3x75497472) -> mlp hidden (reused)
#define OFF_OUT1   379584512u            // out1 f32, 150994944 B
#define WS_NEEDED  530579456u

// ---- single-instruction XOR butterfly via ds_swizzle (group-of-32 mode) ----
template<int XM>
__device__ __forceinline__ float swz_xor_f(float x) {
  return __int_as_float(
      __builtin_amdgcn_ds_swizzle(__float_as_int(x), (XM << 10) | 0x1f));
}
__device__ __forceinline__ float red_add32(float s) {
  s += swz_xor_f<16>(s); s += swz_xor_f<8>(s); s += swz_xor_f<4>(s);
  s += swz_xor_f<2>(s);  s += swz_xor_f<1>(s); return s;
}
__device__ __forceinline__ float red_add16(float s) {
  s += swz_xor_f<8>(s); s += swz_xor_f<4>(s);
  s += swz_xor_f<2>(s); s += swz_xor_f<1>(s); return s;
}
__device__ __forceinline__ float red_max16(float s) {
  s = fmaxf(s, swz_xor_f<8>(s)); s = fmaxf(s, swz_xor_f<4>(s));
  s = fmaxf(s, swz_xor_f<2>(s)); s = fmaxf(s, swz_xor_f<1>(s)); return s;
}

__device__ __forceinline__ size_t xidx(int row, int col) {
  int win = row >> 6, tok = row & 63;
  int b = win / 576, wr = win % 576;
  int hh = (wr / 24) * 8 + (tok >> 3);
  int ww = (wr % 24) * 8 + (tok & 7);
  return (((size_t)b * CDIM + col) * HIMG + hh) * WIMG + ww;
}

// ---- pack f32 weight [K][N] into k-tile-blocked WMMA-B layout:
//      dst[(kt*N + n)*16 + kl] = {w[2*(kt*16+kl)][n], w[2*(kt*16+kl)+1][n]}
__global__ void pack_w(const float* __restrict__ src, unsigned int* __restrict__ dst,
                       int N, int total) {
  int i = blockIdx.x * blockDim.x + threadIdx.x;
  if (i >= total) return;
  int kl = i & 15;
  int t2 = i >> 4;
  int n = t2 % N;
  int kt = t2 / N;
  int kk = kt * 16 + kl;
  union { _Float16 f[2]; unsigned int u; } cv;
  cv.f[0] = (_Float16)src[(size_t)(2 * kk) * N + n];
  cv.f[1] = (_Float16)src[(size_t)(2 * kk + 1) * N + n];
  dst[i] = cv.u;
}

__global__ void zero_pool(float* p) { p[blockIdx.x * 256 + threadIdx.x] = 0.f; }

// ---- precompute attention bias table btab[head][n][m] (f32, 128 KB, L2-resident) ----
__global__ void make_btab(const float* __restrict__ relb, float* __restrict__ btab) {
  int i = blockIdx.x * 256 + threadIdx.x;   // 8*64*64 = 32768
  int nn = (i >> 6) & 63, mm = i & 63, head = i >> 12;
  int idx = ((nn >> 3) - (mm >> 3) + 7) * 15 + ((nn & 7) - (mm & 7) + 7);
  btab[i] = relb[idx * NHEAD + head];
}

// ---- window partition + LayerNorm1 -> h f16 [NWIN*64][256] ----
__global__ __launch_bounds__(256) void ln1_win(const float* __restrict__ x,
    const float* __restrict__ g1, const float* __restrict__ b1,
    unsigned short* __restrict__ hbuf) {
  __shared__ float sm[64][CDIM];   // 64 KB: one window, token-major
  int win = blockIdx.x;
  int c = threadIdx.x;
  int b = win / 576, wr = win % 576;
  int h0 = (wr / 24) * 8, w0 = (wr % 24) * 8;
  const float* xp = x + ((size_t)b * CDIM + c) * HWIMG;
  for (int rh = 0; rh < 8; ++rh) {
    const float4* rp = (const float4*)(xp + (h0 + rh) * WIMG + w0);
    float4 f0 = rp[0], f1 = rp[1];
    sm[rh * 8 + 0][c] = f0.x; sm[rh * 8 + 1][c] = f0.y;
    sm[rh * 8 + 2][c] = f0.z; sm[rh * 8 + 3][c] = f0.w;
    sm[rh * 8 + 4][c] = f1.x; sm[rh * 8 + 5][c] = f1.y;
    sm[rh * 8 + 6][c] = f1.z; sm[rh * 8 + 7][c] = f1.w;
  }
  __syncthreads();
  int wv = threadIdx.x >> 5, l = threadIdx.x & 31;
  _Float16* hb = (_Float16*)hbuf;
  for (int tt = 0; tt < 8; ++tt) {
    int t = wv * 8 + tt;
    float v[8]; float s = 0.f;
    #pragma unroll
    for (int i = 0; i < 8; ++i) { v[i] = sm[t][l + 32 * i]; s += v[i]; }
    s = red_add32(s);
    float mean = s * (1.0f / 256.0f);
    float q = 0.f;
    #pragma unroll
    for (int i = 0; i < 8; ++i) { float d = v[i] - mean; q += d * d; }
    q = red_add32(q);
    float rs = rsqrtf(q * (1.0f / 256.0f) + 1e-5f);
    size_t base = ((size_t)win * 64 + t) * CDIM;
    #pragma unroll
    for (int i = 0; i < 8; ++i) {
      int cc = l + 32 * i;
      hb[base + cc] = (_Float16)((v[i] - mean) * rs * g1[cc] + b1[cc]);
    }
  }
}

// ---- generic WMMA GEMM: block = 8 waves x (16M x 64N), grid (M/128, N/64) ----
// B panel staged into LDS with global_load_async_to_lds_b128 (ASYNCcnt path),
// shared by all 8 waves; fragments read back as 2x ds_load_b128.
template<int KD, int ND, int MODE>
__global__ __launch_bounds__(256) void gemm_wmma(
    const unsigned short* __restrict__ Ah, const unsigned int* __restrict__ Bp,
    const float* __restrict__ bias,
    unsigned short* __restrict__ o16a, unsigned short* __restrict__ o16b,
    unsigned short* __restrict__ o16c, float* __restrict__ o32,
    const float* __restrict__ resx, const float* __restrict__ gvec) {
  __shared__ unsigned int bsm[8192];   // 32 KB: 8 k-tiles x 64 cols x 16 dwords
  int tid = threadIdx.x;
  int wv = tid >> 5;
  int l = tid & 31;
  int l15 = l & 15, hf = l >> 4;
  int mBase = blockIdx.x * 128 + wv * 16;
  int nBase = blockIdx.y * 64;
  const unsigned int* arow =
      (const unsigned int*)Ah + (size_t)(mBase + l15) * (KD / 2);
  unsigned lbase = (unsigned)(uintptr_t)(&bsm[0]);
  v8f zero = {0.f, 0.f, 0.f, 0.f, 0.f, 0.f, 0.f, 0.f};
  v8f acc[4];
  #pragma unroll
  for (int nt = 0; nt < 4; ++nt) acc[nt] = zero;

  const int NCH = KD / 256;   // chunks of 8 k-tiles (KD is 256 or 1024)
  for (int ch = 0; ch < NCH; ++ch) {
    // async-stage this chunk's B panel (8 x 4KB contiguous pieces)
    #pragma unroll
    for (int i = 0; i < 8; ++i) {
      int kt = ch * 8 + i;
      const unsigned int* gp = Bp + ((size_t)kt * ND + nBase) * 16 + tid * 4;
      unsigned loff = lbase + (unsigned)((i * 1024 + tid * 4) * 4);
      asm volatile("global_load_async_to_lds_b128 %0, %1, off"
                   :: "v"(loff), "v"(gp) : "memory");
    }
    asm volatile("s_wait_asynccnt 0x0" ::: "memory");
    __syncthreads();

    #pragma unroll
    for (int i = 0; i < 8; ++i) {
      int kt = ch * 8 + i;
      FragH a;
      const unsigned int* ap = arow + kt * 16 + 8 * hf;
      #pragma unroll
      for (int v = 0; v < 8; ++v) a.u[v] = ap[v];
      #pragma unroll
      for (int nt = 0; nt < 4; ++nt) {
        FragH bF;
        const unsigned int* lp = &bsm[i * 1024 + (nt * 16 + l15) * 16 + 8 * hf];
        #pragma unroll
        for (int v = 0; v < 8; ++v) bF.u[v] = lp[v];
        acc[nt] = __builtin_amdgcn_wmma_f32_16x16x32_f16(
            false, a.h, false, bF.h, (short)0, acc[nt], false, false);
      }
    }
    __syncthreads();
  }

  #pragma unroll
  for (int nt = 0; nt < 4; ++nt) {
    #pragma unroll
    for (int r = 0; r < 8; ++r) {
      int row = mBase + r + 8 * hf;
      int col = nBase + nt * 16 + l15;
      float v = acc[nt][r] + bias[col];
      if (MODE == 0) {
        int which = col >> 8, hc = col & 255;
        int head = hc >> 5, d = hc & 31;
        int win = row >> 6, tok = row & 63;
        size_t hb = (size_t)(win * NHEAD + head);
        if (which == 0)
          ((_Float16*)o16a)[(hb * 64 + tok) * 32 + d] =
              (_Float16)(v * 0.17677669529663687f);   // 1/sqrt(32)
        else if (which == 1)
          ((_Float16*)o16b)[(hb * 64 + tok) * 32 + d] = (_Float16)v;
        else
          ((_Float16*)o16c)[(hb * 32 + d) * 64 + tok] = (_Float16)v;  // v^T
      } else if (MODE == 1) {
        o32[(size_t)row * CDIM + col] = v + resx[xidx(row, col)];
      } else if (MODE == 2) {
        float gl = 0.5f * v * (1.0f + erff(v * 0.70710678118f));
        ((_Float16*)o16a)[(size_t)row * ND + col] = (_Float16)gl;
      } else {
        int b = (row >> 6) / 576;
        float res = resx[(size_t)row * CDIM + col] * gvec[b * CDIM + col];
        o32[xidx(row, col)] = v + res;
      }
    }
  }
}

// ---- windowed attention: block = (window, 2 heads); wave = (head, 16-query tile) ----
__global__ __launch_bounds__(256) void attn_win(
    const unsigned short* __restrict__ qb, const unsigned short* __restrict__ kb,
    const unsigned short* __restrict__ vb, const float* __restrict__ btab,
    unsigned short* __restrict__ ob) {
  __shared__ _Float16 pls[8][16][64];   // per-wave softmaxed probabilities
  int win = blockIdx.x >> 2, hg = blockIdx.x & 3;
  int wv = threadIdx.x >> 5, l = threadIdx.x & 31;
  int l15 = l & 15, hf = l >> 4;
  int head = hg * 2 + (wv >> 2);
  int mt = wv & 3;
  size_t hbase = (size_t)(win * NHEAD + head);
  const unsigned int* qw = (const unsigned int*)qb + hbase * 64 * 16; // [64][16] dw
  const unsigned int* kw = (const unsigned int*)kb + hbase * 64 * 16;
  const unsigned int* vw = (const unsigned int*)vb + hbase * 32 * 32; // vT [32][32] dw
  v8f zero = {0.f, 0.f, 0.f, 0.f, 0.f, 0.f, 0.f, 0.f};

  FragH aq;
  {
    const unsigned int* p = qw + (mt * 16 + l15) * 16 + 8 * hf;
    #pragma unroll
    for (int v = 0; v < 8; ++v) aq.u[v] = p[v];
  }
  v8f sc[4];
  #pragma unroll
  for (int nt = 0; nt < 4; ++nt) {
    FragH bk;
    const unsigned int* p = kw + (nt * 16 + l15) * 16 + 8 * hf;
    #pragma unroll
    for (int v = 0; v < 8; ++v) bk.u[v] = p[v];
    sc[nt] = __builtin_amdgcn_wmma_f32_16x16x32_f16(
        false, aq.h, false, bk.h, (short)0, zero, false, false);
  }
  // relative position bias from precomputed table (coalesced affine loads)
  {
    const float* bt = btab + ((size_t)head * 64 + mt * 16 + 8 * hf) * 64 + l15;
    #pragma unroll
    for (int nt = 0; nt < 4; ++nt)
      #pragma unroll
      for (int r = 0; r < 8; ++r)
        sc[nt][r] += bt[r * 64 + nt * 16];
  }
  // row softmax: each row lives in one 16-lane half across the 4 tiles
  #pragma unroll
  for (int r = 0; r < 8; ++r) {
    float mx = fmaxf(fmaxf(sc[0][r], sc[1][r]), fmaxf(sc[2][r], sc[3][r]));
    mx = red_max16(mx);
    float sum = 0.f;
    #pragma unroll
    for (int nt = 0; nt < 4; ++nt) {
      float e = __expf(sc[nt][r] - mx);
      sc[nt][r] = e; sum += e;
    }
    sum = red_add16(sum);
    float inv = 1.0f / sum;
    #pragma unroll
    for (int nt = 0; nt < 4; ++nt) sc[nt][r] *= inv;
  }
  // relayout C(lane=key) -> A(lane=query) through LDS
  #pragma unroll
  for (int nt = 0; nt < 4; ++nt)
    #pragma unroll
    for (int r = 0; r < 8; ++r)
      pls[wv][r + 8 * hf][nt * 16 + l15] = (_Float16)sc[nt][r];
  __syncthreads();

  const unsigned int* pw = (const unsigned int*)&pls[wv][0][0]; // [16][32] dw
  FragH ap2[2];
  #pragma unroll
  for (int kt = 0; kt < 2; ++kt) {
    const unsigned int* p = pw + l15 * 32 + kt * 16 + 8 * hf;
    #pragma unroll
    for (int v = 0; v < 8; ++v) ap2[kt].u[v] = p[v];
  }
  _Float16* op = (_Float16*)ob;
  #pragma unroll
  for (int dt = 0; dt < 2; ++dt) {
    v8f oacc = zero;
    #pragma unroll
    for (int kt = 0; kt < 2; ++kt) {
      FragH bv;
      const unsigned int* p = vw + (dt * 16 + l15) * 32 + kt * 16 + 8 * hf;
      #pragma unroll
      for (int v = 0; v < 8; ++v) bv.u[v] = p[v];
      oacc = __builtin_amdgcn_wmma_f32_16x16x32_f16(
          false, ap2[kt].h, false, bv.h, (short)0, oacc, false, false);
    }
    #pragma unroll
    for (int r = 0; r < 8; ++r) {
      int row = win * 64 + mt * 16 + r + 8 * hf;
      int col = head * 32 + dt * 16 + l15;
      op[(size_t)row * CDIM + col] = (_Float16)oacc[r];
    }
  }
}

// ---- spatial mean pooling (stage 1: per-block partials + atomicAdd) ----
__global__ __launch_bounds__(256) void pool_sum(const float* __restrict__ out1,
                                                float* __restrict__ pooled) {
  int r0 = blockIdx.x * 256;
  int b = r0 / HWIMG;
  int c = threadIdx.x;
  float s = 0.f;
  for (int i = 0; i < 256; ++i) s += out1[(size_t)(r0 + i) * CDIM + c];
  atomicAdd(&pooled[b * CDIM + c], s);
}

// ---- SE gate: g = sigmoid(relu(mean @ w1 + b1) @ w2 + b2) ----
__global__ __launch_bounds__(256) void se_gate(const float* __restrict__ pooled,
    const float* __restrict__ w1, const float* __restrict__ b1,
    const float* __restrict__ w2, const float* __restrict__ b2,
    float* __restrict__ g) {
  __shared__ float pm[CDIM];
  __shared__ float hid[16];
  int b = blockIdx.x, c = threadIdx.x;
  pm[c] = pooled[b * CDIM + c] * (1.0f / 36864.0f);
  __syncthreads();
  if (c < 16) {
    float s = b1[c];
    for (int i = 0; i < CDIM; ++i) s += pm[i] * w1[i * 16 + c];
    hid[c] = fmaxf(s, 0.f);
  }
  __syncthreads();
  float s = b2[c];
  for (int j = 0; j < 16; ++j) s += hid[j] * w2[j * CDIM + c];
  g[b * CDIM + c] = 1.0f / (1.0f + __expf(-s));
}

// ---- xc = out1 * g; h3 = LN3(xc) f16 ----
__global__ __launch_bounds__(256) void ln3_scale(const float* __restrict__ out1,
    const float* __restrict__ g, const float* __restrict__ n3g,
    const float* __restrict__ n3b, unsigned short* __restrict__ h3) {
  int wv = threadIdx.x >> 5, l = threadIdx.x & 31;
  int t = blockIdx.x * 8 + wv;
  int b = t / HWIMG;
  const float* rowp = out1 + (size_t)t * CDIM;
  const float* gp = g + b * CDIM;
  float v[8]; float s = 0.f;
  #pragma unroll
  for (int i = 0; i < 8; ++i) {
    int c = l + 32 * i;
    v[i] = rowp[c] * gp[c]; s += v[i];
  }
  s = red_add32(s);
  float mean = s * (1.0f / 256.0f);
  float q = 0.f;
  #pragma unroll
  for (int i = 0; i < 8; ++i) { float d = v[i] - mean; q += d * d; }
  q = red_add32(q);
  float rs = rsqrtf(q * (1.0f / 256.0f) + 1e-5f);
  _Float16* hp = (_Float16*)h3;
  #pragma unroll
  for (int i = 0; i < 8; ++i) {
    int c = l + 32 * i;
    hp[(size_t)t * CDIM + c] = (_Float16)((v[i] - mean) * rs * n3g[c] + n3b[c]);
  }
}

extern "C" void kernel_launch(void* const* d_in, const int* in_sizes, int n_in,
                              void* d_out, int out_size, void* d_ws, size_t ws_size,
                              hipStream_t stream) {
  if (ws_size < (size_t)WS_NEEDED) return;
  const float* x      = (const float*)d_in[0];
  const float* n1_g   = (const float*)d_in[1];
  const float* n1_b   = (const float*)d_in[2];
  const float* qkv_w  = (const float*)d_in[3];
  const float* qkv_b  = (const float*)d_in[4];
  const float* proj_w = (const float*)d_in[5];
  const float* proj_b = (const float*)d_in[6];
  const float* rel_b  = (const float*)d_in[7];
  const float* se_w1  = (const float*)d_in[8];
  const float* se_b1  = (const float*)d_in[9];
  const float* se_w2  = (const float*)d_in[10];
  const float* se_b2  = (const float*)d_in[11];
  const float* n3_g   = (const float*)d_in[12];
  const float* n3_b   = (const float*)d_in[13];
  const float* mlp_w1 = (const float*)d_in[14];
  const float* mlp_b1 = (const float*)d_in[15];
  const float* mlp_w2 = (const float*)d_in[16];
  const float* mlp_b2 = (const float*)d_in[17];

  char* ws = (char*)d_ws;
  unsigned int* wqkv  = (unsigned int*)(ws + OFF_WQKV);
  unsigned int* wproj = (unsigned int*)(ws + OFF_WPROJ);
  unsigned int* w1p   = (unsigned int*)(ws + OFF_W1);
  unsigned int* w2p   = (unsigned int*)(ws + OFF_W2);
  float* pooled       = (float*)(ws + OFF_POOL);
  float* gbuf         = (float*)(ws + OFF_G);
  float* btab         = (float*)(ws + OFF_BTAB);
  unsigned short* hbuf = (unsigned short*)(ws + OFF_H);         // h -> o -> h3
  unsigned short* qb   = (unsigned short*)(ws + OFF_BIG);
  unsigned short* kb   = (unsigned short*)(ws + OFF_BIG + 75497472u);
  unsigned short* vb   = (unsigned short*)(ws + OFF_BIG + 150994944u);
  unsigned short* mbuf = (unsigned short*)(ws + OFF_BIG);       // reuses q/k/v
  float* out1          = (float*)(ws + OFF_OUT1);
  float* outp          = (float*)d_out;

  pack_w<<<dim3(384), 256, 0, stream>>>(qkv_w, wqkv, 768, 98304);
  pack_w<<<dim3(128), 256, 0, stream>>>(proj_w, wproj, 256, 32768);
  pack_w<<<dim3(512), 256, 0, stream>>>(mlp_w1, w1p, 1024, 131072);
  pack_w<<<dim3(512), 256, 0, stream>>>(mlp_w2, w2p, 256, 131072);
  zero_pool<<<dim3(4), 256, 0, stream>>>(pooled);
  make_btab<<<dim3(128), 256, 0, stream>>>(rel_b, btab);

  ln1_win<<<dim3(NWIN), 256, 0, stream>>>(x, n1_g, n1_b, hbuf);
  gemm_wmma<256, 768, 0><<<dim3(1152, 12), 256, 0, stream>>>(
      hbuf, wqkv, qkv_b, qb, kb, vb, nullptr, nullptr, nullptr);
  attn_win<<<dim3(NWIN * 4), 256, 0, stream>>>(qb, kb, vb, btab, hbuf /*o reuses h*/);
  gemm_wmma<256, 256, 1><<<dim3(1152, 4), 256, 0, stream>>>(
      hbuf, wproj, proj_b, nullptr, nullptr, nullptr, out1, x, nullptr);
  pool_sum<<<dim3(576), 256, 0, stream>>>(out1, pooled);
  se_gate<<<dim3(4), 256, 0, stream>>>(pooled, se_w1, se_b1, se_w2, se_b2, gbuf);
  ln3_scale<<<dim3(T_TOK / 8), 256, 0, stream>>>(out1, gbuf, n3_g, n3_b, hbuf /*h3*/);
  gemm_wmma<256, 1024, 2><<<dim3(1152, 16), 256, 0, stream>>>(
      hbuf, w1p, mlp_b1, mbuf, nullptr, nullptr, nullptr, nullptr, nullptr);
  gemm_wmma<1024, 256, 3><<<dim3(1152, 4), 256, 0, stream>>>(
      mbuf, w2p, mlp_b2, nullptr, nullptr, nullptr, outp, out1, gbuf);
}
